// IndexedConv_76965813944388
// MI455X (gfx1250) — matrix operand order repaired and checked by reading, not compiled, not run on hardware
//
#include <hip/hip_runtime.h>
#include <hip/hip_bf16.h>

typedef __attribute__((ext_vector_type(16))) __bf16 v16bf;
typedef __attribute__((ext_vector_type(8)))  float  v8f;
typedef __attribute__((ext_vector_type(4)))  int    v4i;
typedef __attribute__((ext_vector_type(8)))  int    v8i;
typedef __attribute__((ext_vector_type(4)))  unsigned int v4u;

typedef unsigned int u32;
typedef unsigned long long u64;

#define N_B   8
#define C_IN  128
#define C_OUT 256
#define KK    7
#define LL    16384
#define LT    64          // l-tile per block
#define KSTEPS 28         // 7 k-taps * (128/32) channel chunks

union FragBF { v16bf v; v4i q[2]; };

// ---------------------------------------------------------------------------
// Prep 1: x (N, C_in, L) f32  ->  xT (N, L, C_in) bf16   (tiled transpose)
// ---------------------------------------------------------------------------
__global__ __launch_bounds__(256) void ic_xpose(const float* __restrict__ x,
                                                __bf16* __restrict__ xT) {
    __shared__ float tile[32][33];
    const int n  = blockIdx.z;
    const int c0 = blockIdx.y * 32;
    const int l0 = blockIdx.x * 32;
    const int t  = threadIdx.x;
    const int lc = t & 31;       // l within tile (coalesced)
    const int cr = t >> 5;       // 0..7
#pragma unroll
    for (int i = 0; i < 4; ++i) {
        const int c = cr + i * 8;
        tile[c][lc] = x[((size_t)n * C_IN + c0 + c) * LL + l0 + lc];
    }
    __syncthreads();
    const int lr = t >> 3;        // 0..31
    const int cc = (t & 7) * 4;   // 0..28 step 4
    __bf16* dst = xT + ((size_t)n * LL + l0 + lr) * C_IN + c0 + cc;
#pragma unroll
    for (int j = 0; j < 4; ++j) dst[j] = (__bf16)tile[cc + j][lr];
}

// ---------------------------------------------------------------------------
// Prep 2: W (C_out, C_in, K) f32 -> Wb[o][k*128+c] bf16 ; idx i64 -> i32
// ---------------------------------------------------------------------------
__global__ __launch_bounds__(256) void ic_prep(const float* __restrict__ w,
                                               const long long* __restrict__ idx,
                                               __bf16* __restrict__ Wb,
                                               int* __restrict__ gidx) {
    const int gid = blockIdx.x * 256 + threadIdx.x;
    if (gid < C_OUT * C_IN * KK) {
        const int o   = gid / (C_IN * KK);
        const int rem = gid % (C_IN * KK);
        const int k   = rem / C_IN;
        const int c   = rem % C_IN;
        Wb[gid] = (__bf16)w[((size_t)o * C_IN + c) * KK + k];
    }
    if (gid < KK * LL) {
        const long long v = idx[gid];
        gidx[gid] = (v < 0) ? -1 : (int)v;
    }
}

// ---------------------------------------------------------------------------
// TDM: DMA one 256x32-bf16 weight tile (row stride 896 elems) into LDS.
// D# per CDNA5 ISA ch.8: group0 {count,lds_addr,global_addr,type=2},
// group1 {data_size=2B, dims, tile dims, stride}; groups 2/3 zero (2-D).
// 6-arg builtin form (clang-23 / therock-10.0 headers).
// ---------------------------------------------------------------------------
__device__ __forceinline__ void tdm_load_wtile(const __bf16* gsrc, u32 lds_off) {
    const u64 ga = (u64)(size_t)gsrc;
    v4u g0 = { 1u,                                     // count=1 (valid user D#)
               lds_off,                                // lds_addr [63:32]
               (u32)ga,                                // global_addr [95:64]
               (u32)((ga >> 32) & 0x01FFFFFFu) | (2u << 30) };  // addr hi | type=2
    v8i g1 = { (int)(1u << 16),        // workgroup_mask=0 | data_size=1 (2 bytes)
               (int)(896u << 16),      // tensor_dim0 lo16  (dim0 = 896)
               (int)(256u << 16),      // dim0 hi16=0 | tensor_dim1 lo16 (=256)
               (int)(32u  << 16),      // dim1 hi16=0 | tile_dim0 = 32
               (int)256,               // tile_dim1 = 256 | tile_dim2 = 0
               (int)896,               // tensor_dim0_stride lo32 = 896
               0, 0 };                 // stride hi | tensor_dim1_stride = 0
    v4i gz4 = { 0, 0, 0, 0 };
    v8i gz8 = { 0, 0, 0, 0, 0, 0, 0, 0 };
    __builtin_amdgcn_tensor_load_to_lds(g0, g1, gz4, gz4, gz8, 0);
}

// ---------------------------------------------------------------------------
// Main GEMM: block = (n, 64-wide l-tile), 8 waves x (32 M x 64 N) each.
// ---------------------------------------------------------------------------
__global__ __launch_bounds__(256) void ic_main(const __bf16* __restrict__ xT,
                                               const __bf16* __restrict__ Wb,
                                               const int* __restrict__ gidx,
                                               const float* __restrict__ bias,
                                               float* __restrict__ out) {
    __shared__ __align__(16) __bf16 As[2][C_OUT * 32];  // 2 x 16 KB (TDM dst)
    __shared__ __align__(16) __bf16 Bs[LT * 32];        // 4 KB gathered tile

    const int n    = blockIdx.y;
    const int l0   = blockIdx.x * LT;
    const int t    = threadIdx.x;
    const int wave = t >> 5;
    const int lane = t & 31;
    const int half = lane >> 4;       // 0 or 1
    const int lr   = lane & 15;
    const int m0   = wave * 32;       // this wave's 32 output channels

    v8f acc[2][4];
#pragma unroll
    for (int mt = 0; mt < 2; ++mt)
#pragma unroll
        for (int nt = 0; nt < 4; ++nt) {
            v8f z = {0.f, 0.f, 0.f, 0.f, 0.f, 0.f, 0.f, 0.f};
            acc[mt][nt] = z;
        }

    // B staging role: 4 threads per gathered row, 16 bytes each
    const int brow = t >> 2;          // 0..63
    const int bseg = t & 3;           // 0..3

    // Kick off TDM for k-step 0 (wave 0 owns the tensor DMA).
    if (wave == 0)
        tdm_load_wtile(Wb, (u32)(size_t)&As[0][0]);

    for (int s = 0; s < KSTEPS; ++s) {
        const int k  = s >> 2;
        const int c0 = (s & 3) << 5;

        // ---- gather masked B tile into LDS (data-dependent -> VGPR path)
        const int gi = gidx[k * LL + l0 + brow];
        v4i bv = {0, 0, 0, 0};
        if (gi >= 0)
            bv = *(const v4i*)(xT + ((size_t)n * LL + gi) * C_IN + c0 + bseg * 8);
        *(v4i*)&Bs[brow * 32 + bseg * 8] = bv;

        // ---- pipeline weight tiles through the TDM (double buffered)
        if (wave == 0) {
            if (s + 1 < KSTEPS) {
                const int kc1 = ((s + 1) >> 2) * C_IN + (((s + 1) & 3) << 5);
                tdm_load_wtile(Wb + kc1, (u32)(size_t)&As[(s + 1) & 1][0]);
                __builtin_amdgcn_s_wait_tensorcnt(1);   // tile s landed
            } else {
                __builtin_amdgcn_s_wait_tensorcnt(0);
            }
        }
        __syncthreads();   // publish A (TDM) + B (ds stores) to all waves

        // ---- fragment loads per ISA 7.12.2 lane layouts
        const __bf16* Ab = &As[s & 1][0];
        FragBF a[2], b[4];
#pragma unroll
        for (int mt = 0; mt < 2; ++mt) {
            const int row = m0 + mt * 16 + lr;
            const v4i* Ap = (const v4i*)&Ab[row * 32 + half * 8];
            a[mt].q[0] = Ap[0];       // K = half*8 + 0..7
            a[mt].q[1] = Ap[2];       // K = 16 + half*8 + 0..7
        }
#pragma unroll
        for (int nt = 0; nt < 4; ++nt) {
            const int row = nt * 16 + lr;
            const v4i* Bp = (const v4i*)&Bs[row * 32 + half * 8];
            b[nt].q[0] = Bp[0];       // K = half*8 + 0..7
            b[nt].q[1] = Bp[2];       // K = 16 + half*8 + 0..7
        }

        // ---- 8 WMMAs: 32x64 tile contribution
#pragma unroll
        for (int nt = 0; nt < 4; ++nt) {
            acc[0][nt] = __builtin_amdgcn_wmma_f32_16x16x32_bf16(
                false, a[0].v, false, b[nt].v, (short)0, acc[0][nt], false, false);
            acc[1][nt] = __builtin_amdgcn_wmma_f32_16x16x32_bf16(
                false, a[1].v, false, b[nt].v, (short)0, acc[1][nt], false, false);
        }
        __syncthreads();   // Bs safe to overwrite next step
    }

    // ---- epilogue: bias + store per C/D layout (VGPR r: M = half*8 + r)
    float bv[2][8];
#pragma unroll
    for (int mt = 0; mt < 2; ++mt)
#pragma unroll
        for (int r = 0; r < 8; ++r)
            bv[mt][r] = bias[m0 + mt * 16 + half * 8 + r];

#pragma unroll
    for (int mt = 0; mt < 2; ++mt)
#pragma unroll
        for (int nt = 0; nt < 4; ++nt) {
            const int l = l0 + nt * 16 + lr;
#pragma unroll
            for (int r = 0; r < 8; ++r) {
                const int o = m0 + mt * 16 + half * 8 + r;
                out[((size_t)n * C_OUT + o) * LL + l] = acc[mt][nt][r] + bv[mt][r];
            }
        }
}

// ---------------------------------------------------------------------------
extern "C" void kernel_launch(void* const* d_in, const int* in_sizes, int n_in,
                              void* d_out, int out_size, void* d_ws, size_t ws_size,
                              hipStream_t stream) {
    const float*     x    = (const float*)d_in[0];        // (8,128,16384) f32
    const long long* idx  = (const long long*)d_in[1];    // (7,16384) i64
    const float*     w    = (const float*)d_in[2];        // (256,128,7) f32
    const float*     bias = (const float*)d_in[3];        // (256,) f32
    float*           out  = (float*)d_out;                // (8,256,16384) f32

    // workspace partition
    char* ws = (char*)d_ws;
    __bf16* xT   = (__bf16*)ws;                               // 33,554,432 B
    __bf16* Wb   = (__bf16*)(ws + (size_t)N_B * LL * C_IN * 2);   // 458,752 B
    int*    gidx = (int*)(ws + (size_t)N_B * LL * C_IN * 2
                             + (size_t)C_OUT * C_IN * KK * 2);    // 458,752 B

    ic_xpose<<<dim3(LL / 32, C_IN / 32, N_B), 256, 0, stream>>>(x, xT);
    ic_prep<<<(C_OUT * C_IN * KK + 255) / 256, 256, 0, stream>>>(w, idx, Wb, gidx);
    ic_main<<<dim3(LL / LT, N_B), 256, 0, stream>>>(xT, Wb, gidx, bias, out);
}
